// DigitConvolutionalModel_46273977647609
// MI455X (gfx1250) — matrix operand order repaired and checked by reading, not compiled, not run on hardware
//
#include <hip/hip_runtime.h>
#include <hip/hip_bf16.h>

typedef __attribute__((ext_vector_type(16))) _Float16 v16h;
typedef __attribute__((ext_vector_type(8)))  _Float16 v8h;
typedef __attribute__((ext_vector_type(8)))  float    v8f;

#define BATCH 16384

// ---------------------------------------------------------------------------
// Convert fp32 weights [N,K] -> fp16 [Npad,Kpad], zero padding.
// ---------------------------------------------------------------------------
__global__ __launch_bounds__(256) void cvt_pad_f16(
    const float* __restrict__ W, _Float16* __restrict__ Wh,
    int N, int K, int Npad, int Kpad)
{
    int total = Npad * Kpad;
    for (int t = blockIdx.x * blockDim.x + threadIdx.x; t < total;
         t += gridDim.x * blockDim.x) {
        int n = t / Kpad;
        int k = t - n * Kpad;
        _Float16 v = (_Float16)0.0f;
        if (n < N && k < K) v = (_Float16)W[(size_t)n * K + k];
        Wh[t] = v;
    }
}

// ---------------------------------------------------------------------------
// 3x3 VALID cross-correlation on 28x28 images -> fp16 [B, 704] (676 real cols,
// cols 676..703 zeroed so GEMM1 can use Kpad=704).
// ---------------------------------------------------------------------------
__global__ __launch_bounds__(256) void conv3x3_kernel(
    const float* __restrict__ x, const float* __restrict__ w,
    _Float16* __restrict__ h0)
{
    const int b = blockIdx.x;
    const float* xb = x + (size_t)b * 784;

    float w00 = w[0], w01 = w[1], w02 = w[2];
    float w10 = w[3], w11 = w[4], w12 = w[5];
    float w20 = w[6], w21 = w[7], w22 = w[8];

    for (int c = threadIdx.x; c < 704; c += blockDim.x) {
        _Float16 out = (_Float16)0.0f;
        if (c < 676) {
            int r  = c / 26;
            int cc = c - r * 26;
            const float* p0 = xb + r * 28 + cc;
            const float* p1 = p0 + 28;
            const float* p2 = p1 + 28;
            float s = p0[0]*w00 + p0[1]*w01 + p0[2]*w02
                    + p1[0]*w10 + p1[1]*w11 + p1[2]*w12
                    + p2[0]*w20 + p2[1]*w21 + p2[2]*w22;
            out = (_Float16)s;
        }
        h0[(size_t)b * 704 + c] = out;
    }
}

// ---------------------------------------------------------------------------
// WMMA GEMM: H[M,N] = act(A[M,K] @ W[N,K]^T + bias), fp16 in, fp16 out.
// Block = 256 threads = 8 waves; wave w owns a (MT*16) x (NT*16) output tile:
//   rows m0 = (blk.x*8+w)*MT*16, cols n0 = blk.y*NT*16.  K multiple of 32.
// Each k-step: MT A-fragments + NT B-fragments feed MT*NT WMMAs, so every
// B fragment is reused MT times and every A fragment NT times from registers.
// ---------------------------------------------------------------------------
union AFrag { v16h v; v8h h[2]; };

template<int MT, int NT, bool RELU>
__global__ __launch_bounds__(256) void gemm_wmma_f16(
    const _Float16* __restrict__ A, const _Float16* __restrict__ W,
    const float* __restrict__ bias, _Float16* __restrict__ H,
    int K, int N)
{
    const int lane = threadIdx.x & 31;
    const int wave = threadIdx.x >> 5;
    const int m0   = (blockIdx.x * 8 + wave) * (MT * 16);
    const int n0   = blockIdx.y * (NT * 16);
    const int row  = lane & 15;     // M row (A frag) / N col (B,C frags)
    const int hi   = lane >> 4;     // half-wave select

    v8f acc[MT][NT];
#pragma unroll
    for (int mt = 0; mt < MT; ++mt)
#pragma unroll
        for (int nt = 0; nt < NT; ++nt)
            acc[mt][nt] = (v8f){0.f,0.f,0.f,0.f,0.f,0.f,0.f,0.f};

    // A fragment base: lane holds row (m0+mt*16+row); low half-wave K..K+7 /
    // K+16..23, high half-wave K+8..15 / K+24..31 (ISA 16-bit A 16x32 layout).
    const _Float16* ap = A + (size_t)(m0 + row) * K + hi * 8;
    // B fragment base: lane holds weight row (n0+nt*16+row); low half-wave
    // K..K+7 / K+16..23, high half-wave K+8..15 / K+24..31 (same as A).
    const _Float16* bp = W + (size_t)(n0 + row) * K + hi * 8;

    for (int k0 = 0; k0 < K; k0 += 32) {
        AFrag a[MT];
#pragma unroll
        for (int mt = 0; mt < MT; ++mt) {
            const _Float16* amt = ap + (size_t)mt * 16 * K + k0;
            a[mt].h[0] = *(const v8h*)(amt);
            a[mt].h[1] = *(const v8h*)(amt + 16);
        }
#pragma unroll
        for (int nt = 0; nt < NT; ++nt) {
            AFrag bf; bf.h[0] = *(const v8h*)(bp + (size_t)nt * 16 * K + k0); bf.h[1] = *(const v8h*)(bp + (size_t)nt * 16 * K + k0 + 16); v16h b = bf.v;
#pragma unroll
            for (int mt = 0; mt < MT; ++mt) {
                acc[mt][nt] = __builtin_amdgcn_wmma_f32_16x16x32_f16(
                    false, a[mt].v, false, b, (short)0, acc[mt][nt], false, false);
            }
        }
    }

    // Epilogue: C/D 16x16 f32 layout — lane n + 16*(m>=8), VGPR r = m&7.
#pragma unroll
    for (int nt = 0; nt < NT; ++nt) {
        const int gn = n0 + nt * 16 + row;
        const float bv = bias[gn];
#pragma unroll
        for (int mt = 0; mt < MT; ++mt) {
#pragma unroll
            for (int r = 0; r < 8; ++r) {
                const int m = m0 + mt * 16 + hi * 8 + r;
                float v = acc[mt][nt][r] + bv;
                if (RELU) v = v > 0.f ? v : 0.f;
                H[(size_t)m * N + gn] = (_Float16)v;
            }
        }
    }
}

// ---------------------------------------------------------------------------
// Final layer: [M,256] @ W4h[16,256]^T + b4 -> fp32 logits [M,10].
// ---------------------------------------------------------------------------
__global__ __launch_bounds__(256) void gemm_wmma_final(
    const _Float16* __restrict__ A, const _Float16* __restrict__ W,
    const float* __restrict__ bias, float* __restrict__ out,
    int K, int Nout)
{
    const int lane = threadIdx.x & 31;
    const int wave = threadIdx.x >> 5;
    const int m0   = (blockIdx.x * 8 + wave) * 16;
    const int row  = lane & 15;
    const int hi   = lane >> 4;

    v8f acc = (v8f){0.f,0.f,0.f,0.f,0.f,0.f,0.f,0.f};

    const _Float16* ap = A + (size_t)(m0 + row) * K + hi * 8;
    const _Float16* bp = W + (size_t)row * K + hi * 8;

    for (int k0 = 0; k0 < K; k0 += 32) {
        AFrag a;
        a.h[0] = *(const v8h*)(ap + k0);
        a.h[1] = *(const v8h*)(ap + k0 + 16);
        AFrag bf; bf.h[0] = *(const v8h*)(bp + k0); bf.h[1] = *(const v8h*)(bp + k0 + 16); v16h b = bf.v;
        acc = __builtin_amdgcn_wmma_f32_16x16x32_f16(
            false, a.v, false, b, (short)0, acc, false, false);
    }

    if (row < Nout) {
        const float bv = bias[row];
#pragma unroll
        for (int r = 0; r < 8; ++r) {
            const int m = m0 + hi * 8 + r;
            out[(size_t)m * Nout + row] = acc[r] + bv;
        }
    }
}

// ---------------------------------------------------------------------------
// Launch
// ---------------------------------------------------------------------------
extern "C" void kernel_launch(void* const* d_in, const int* in_sizes, int n_in,
                              void* d_out, int out_size, void* d_ws, size_t ws_size,
                              hipStream_t stream) {
    const float* x      = (const float*)d_in[0];
    const float* conv_w = (const float*)d_in[1];
    const float* W1 = (const float*)d_in[2];  const float* b1 = (const float*)d_in[3];
    const float* W2 = (const float*)d_in[4];  const float* b2 = (const float*)d_in[5];
    const float* W3 = (const float*)d_in[6];  const float* b3 = (const float*)d_in[7];
    const float* W4 = (const float*)d_in[8];  const float* b4 = (const float*)d_in[9];
    float* logits = (float*)d_out;

    // Workspace carve-up (256B aligned slices).
    size_t off = 0;
    auto carve = [&](size_t bytes) {
        void* p = (char*)d_ws + off;
        off += (bytes + 255) & ~(size_t)255;
        return p;
    };
    _Float16* h0  = (_Float16*)carve((size_t)BATCH * 704 * 2);
    _Float16* h1  = (_Float16*)carve((size_t)BATCH * 1024 * 2);
    _Float16* h2  = (_Float16*)carve((size_t)BATCH * 512 * 2);
    _Float16* h3  = (_Float16*)carve((size_t)BATCH * 256 * 2);
    _Float16* W1h = (_Float16*)carve((size_t)1024 * 704 * 2);
    _Float16* W2h = (_Float16*)carve((size_t)512 * 1024 * 2);
    _Float16* W3h = (_Float16*)carve((size_t)256 * 512 * 2);
    _Float16* W4h = (_Float16*)carve((size_t)16 * 256 * 2);

    // 1) Weight conversion fp32 -> fp16 (with padding).
    auto cvt = [&](const float* w, _Float16* wh, int N, int K, int Np, int Kp) {
        int total = Np * Kp;
        int blocks = (total + 255) / 256;
        if (blocks > 4096) blocks = 4096;
        cvt_pad_f16<<<blocks, 256, 0, stream>>>(w, wh, N, K, Np, Kp);
    };
    cvt(W1, W1h, 1024, 676, 1024, 704);
    cvt(W2, W2h,  512, 1024,  512, 1024);
    cvt(W3, W3h,  256,  512,  256,  512);
    cvt(W4, W4h,   10,  256,   16,  256);

    // 2) Conv front-end -> h0 fp16 [B, 704].
    conv3x3_kernel<<<BATCH, 256, 0, stream>>>(x, conv_w, h0);

    // 3) GEMM chain: 32x64 tile per wave, 8 waves/block => 256 rows/block,
    //    grid.x = 16384/256 = 64.
    gemm_wmma_f16<2, 4, true><<<dim3(64, 16), 256, 0, stream>>>(h0, W1h, b1, h1,  704, 1024);
    gemm_wmma_f16<2, 4, true><<<dim3(64,  8), 256, 0, stream>>>(h1, W2h, b2, h2, 1024,  512);
    gemm_wmma_f16<2, 4, true><<<dim3(64,  4), 256, 0, stream>>>(h2, W3h, b3, h3,  512,  256);

    // 4) Final logits fp32 [B, 10].
    gemm_wmma_final<<<dim3(128, 1), 256, 0, stream>>>(h3, W4h, b4, logits, 256, 10);
}
